// MultiHeadAttention_4415226380458
// MI455X (gfx1250) — matrix operand order repaired and checked by reading, not compiled, not run on hardware
//
#include <hip/hip_runtime.h>
#include <stdint.h>

typedef unsigned short ushort_t;
typedef __attribute__((ext_vector_type(16))) __bf16    v16bf;
typedef __attribute__((ext_vector_type(8)))  float     v8f;
typedef __attribute__((ext_vector_type(4)))  unsigned  v4u;
typedef __attribute__((ext_vector_type(4)))  int       v4i_t;

#define BDIM 4
#define TDIM 2048
#define CDIM 2048
#define HDIM 16
#define DDIM 128
#define NQKV 6144

#if __has_builtin(__builtin_amdgcn_global_load_async_to_lds_b128)
#define USE_ASYNC 1
typedef __attribute__((address_space(1))) v4i_t glb_v4i;
typedef __attribute__((address_space(3))) v4i_t lds_v4i;
#endif

__device__ __forceinline__ ushort_t f2bf(float f) {
  unsigned u = __float_as_uint(f);
  u += 0x7fffu + ((u >> 16) & 1u);          // round-to-nearest-even
  return (ushort_t)(u >> 16);
}
__device__ __forceinline__ float bf2f(ushort_t h) {
  return __uint_as_float(((unsigned)h) << 16);
}
__device__ __forceinline__ unsigned pack2(float lo, float hi) {
  return ((unsigned)f2bf(lo)) | (((unsigned)f2bf(hi)) << 16);
}

// ---- async global -> LDS 16B copy (CDNA5 ASYNCcnt path) ------------------
__device__ __forceinline__ void cp16(ushort_t* lds, const ushort_t* g) {
#ifdef USE_ASYNC
  __builtin_amdgcn_global_load_async_to_lds_b128(
      (glb_v4i*)g, (lds_v4i*)lds, 0, 0);
#else
  *(v4u*)lds = *(const v4u*)g;
#endif
}
__device__ __forceinline__ void wait_async() {
#ifdef USE_ASYNC
#if __has_builtin(__builtin_amdgcn_s_wait_asynccnt)
  __builtin_amdgcn_s_wait_asynccnt(0);
#else
  asm volatile("s_wait_asynccnt 0x0" ::: "memory");
#endif
#endif
}

// ---- WMMA helpers --------------------------------------------------------
// A fragment (16x32 bf16): lane m = row; two contiguous 16B runs at
// (base + k0) and (base + k0 + 16), k0 = (lane>=16 ? 8 : 0).
__device__ __forceinline__ v16bf ldfragA(const ushort_t* rowbase, int k0) {
  union { v16bf v; v4u q[2]; } u;
  u.q[0] = *(const v4u*)(rowbase + k0);
  u.q[1] = *(const v4u*)(rowbase + k0 + 16);
  return u.v;
}
// B fragment (32x16 bf16): lane n = col; two contiguous 16B runs at
// (base + k0/2) and (base + k0/2 + 16), k0 = (lane>=16 ? 16 : 0).
__device__ __forceinline__ v16bf ldfragB(const ushort_t* rowbase, int k0) {
  union { v16bf v; v4u q[2]; } u;
  u.q[0] = *(const v4u*)(rowbase + k0 / 2);
  u.q[1] = *(const v4u*)(rowbase + k0 / 2 + 16);
  return u.v;
}
__device__ __forceinline__ v8f wmma_bf16(v16bf a, v16bf b, v8f c) {
  return __builtin_amdgcn_wmma_f32_16x16x32_bf16(false, a, false, b,
                                                 (short)0, c, false, false);
}

// ---- 16-lane butterfly reductions (VALU-only via v_permlane16_b32) -------
#if __has_builtin(__builtin_amdgcn_permlane16)
__device__ __forceinline__ float xor16(float v, unsigned s1, unsigned s2) {
  unsigned i = __float_as_uint(v);
  return __uint_as_float(__builtin_amdgcn_permlane16(i, i, s1, s2, false, false));
}
__device__ __forceinline__ float redmax16(float v) {
  v = fmaxf(v, xor16(v, 0x67452301u, 0xEFCDAB89u));   // xor 1
  v = fmaxf(v, xor16(v, 0x54761032u, 0xDCFE98BAu));   // xor 2
  v = fmaxf(v, xor16(v, 0x32107654u, 0xBA98FEDCu));   // xor 4
  v = fmaxf(v, xor16(v, 0xFEDCBA98u, 0x76543210u));   // xor 8
  return v;
}
__device__ __forceinline__ float redsum16(float v) {
  v += xor16(v, 0x67452301u, 0xEFCDAB89u);
  v += xor16(v, 0x54761032u, 0xDCFE98BAu);
  v += xor16(v, 0x32107654u, 0xBA98FEDCu);
  v += xor16(v, 0xFEDCBA98u, 0x76543210u);
  return v;
}
#else
__device__ __forceinline__ float redmax16(float v) {
#pragma unroll
  for (int m = 1; m < 16; m <<= 1) v = fmaxf(v, __shfl_xor(v, m, 32));
  return v;
}
__device__ __forceinline__ float redsum16(float v) {
#pragma unroll
  for (int m = 1; m < 16; m <<= 1) v += __shfl_xor(v, m, 32);
  return v;
}
#endif

// ---- fp32 -> bf16 conversion (8 elems / thread) --------------------------
__global__ void cvt_f32_bf16(const float* __restrict__ in,
                             ushort_t* __restrict__ out, int n8) {
  int i = blockIdx.x * blockDim.x + threadIdx.x;
  if (i < n8) {
    const float4* p = (const float4*)in + (size_t)i * 2;
    float4 a = p[0], b = p[1];
    v4u o;
    o.x = pack2(a.x, a.y); o.y = pack2(a.z, a.w);
    o.z = pack2(b.x, b.y); o.w = pack2(b.z, b.w);
    *(v4u*)&out[(size_t)i * 8] = o;
  }
}

// ---- Tiled WMMA GEMM: C[M,N] = A[M,K] * B[N,K]^T + bias ------------------
// 128x128 block tile, K-step 32, double-buffered LDS, async pipelined.
#define SSTR 40   // 32 + 8 pad, 80B row stride (multiple of 16B)
template <bool OUT_BF16>
__global__ __launch_bounds__(256, 2)
void gemm_bf16_nt(const ushort_t* __restrict__ A, const ushort_t* __restrict__ Bm,
                  const float* __restrict__ bias, void* __restrict__ Cout,
                  int M, int N, int K) {
  __shared__ __align__(16) ushort_t As[2][128 * SSTR];
  __shared__ __align__(16) ushort_t Bs[2][128 * SSTR];

  const int tid  = threadIdx.x;
  const int wave = tid >> 5, lane = tid & 31;
  const int hl = lane >> 4, ln = lane & 15;
  const int bm = blockIdx.y, bn = blockIdx.x;
  const int waveM = (wave >> 2) * 64;
  const int waveN = (wave & 3) * 32;

  const ushort_t* Arow = A + (size_t)(bm * 128) * K;
  const ushort_t* Brow = Bm + (size_t)(bn * 128) * K;
  const int r0 = tid >> 2,        cc0 = (tid & 3) * 8;
  const int r1 = (tid + 256) >> 2, cc1 = (tid & 3) * 8;

  v8f acc[4][2];
#pragma unroll
  for (int i = 0; i < 4; ++i)
#pragma unroll
    for (int j = 0; j < 2; ++j) acc[i][j] = (v8f)0.0f;

  // prologue: issue first tile
  cp16(&As[0][r0 * SSTR + cc0], &Arow[(size_t)r0 * K + cc0]);
  cp16(&Bs[0][r0 * SSTR + cc0], &Brow[(size_t)r0 * K + cc0]);
  cp16(&As[0][r1 * SSTR + cc1], &Arow[(size_t)r1 * K + cc1]);
  cp16(&Bs[0][r1 * SSTR + cc1], &Brow[(size_t)r1 * K + cc1]);

  int ib = 0;
  for (int k0 = 0; k0 < K; k0 += 32, ib ^= 1) {
    wait_async();
    __syncthreads();
    if (k0 + 32 < K) {           // issue next tile into other buffer
      int kn = k0 + 32;
      cp16(&As[ib ^ 1][r0 * SSTR + cc0], &Arow[(size_t)r0 * K + kn + cc0]);
      cp16(&Bs[ib ^ 1][r0 * SSTR + cc0], &Brow[(size_t)r0 * K + kn + cc0]);
      cp16(&As[ib ^ 1][r1 * SSTR + cc1], &Arow[(size_t)r1 * K + kn + cc1]);
      cp16(&Bs[ib ^ 1][r1 * SSTR + cc1], &Brow[(size_t)r1 * K + kn + cc1]);
    }
    // batched fragment loads, then back-to-back WMMAs
    v16bf af[4];
#pragma unroll
    for (int mt = 0; mt < 4; ++mt)
      af[mt] = ldfragA(&As[ib][(waveM + mt * 16 + ln) * SSTR], hl * 8);
    v16bf bf0 = ldfragB(&Bs[ib][(waveN + ln) * SSTR], hl * 16);
    v16bf bf1 = ldfragB(&Bs[ib][(waveN + 16 + ln) * SSTR], hl * 16);
#pragma unroll
    for (int mt = 0; mt < 4; ++mt) {
      acc[mt][0] = wmma_bf16(af[mt], bf0, acc[mt][0]);
      acc[mt][1] = wmma_bf16(af[mt], bf1, acc[mt][1]);
    }
  }

#pragma unroll
  for (int mt = 0; mt < 4; ++mt)
#pragma unroll
    for (int nt = 0; nt < 2; ++nt)
#pragma unroll
      for (int r = 0; r < 8; ++r) {
        int row = bm * 128 + waveM + mt * 16 + r + hl * 8;
        int col = bn * 128 + waveN + nt * 16 + ln;
        float v = acc[mt][nt][r] + bias[col];
        if (OUT_BF16) ((ushort_t*)Cout)[(size_t)row * N + col] = f2bf(v);
        else          ((float*)Cout)[(size_t)row * N + col]    = v;
      }
}

// ---- RoPE + layout: qkv[B,T,3,H,D] -> q,k [B,H,T,D], v -> vt [B,H,D,T] ---
__global__ void rope_rearrange(const ushort_t* __restrict__ qkv,
                               ushort_t* __restrict__ q, ushort_t* __restrict__ k,
                               ushort_t* __restrict__ vt) {
  int i = blockIdx.x * blockDim.x + threadIdx.x;   // B*H*T*(D/2)
  int j = i & 63;
  int t = (i >> 6) & (TDIM - 1);
  int h = (i >> 17) & (HDIM - 1);
  int b = i >> 21;

  size_t base = (size_t)(b * TDIM + t) * NQKV + h * DDIM;
  float theta = __expf(-(float)j * 0.14391156f);   // ln(10000)/64
  float s, c;
  __sincosf((float)t * theta, &s, &c);

  float x1 = bf2f(qkv[base + 2 * j]), x2 = bf2f(qkv[base + 2 * j + 1]);
  size_t qo = ((size_t)(b * HDIM + h) * TDIM + t) * DDIM + 2 * j;
  q[qo]     = f2bf(x1 * c - x2 * s);
  q[qo + 1] = f2bf(x1 * s + x2 * c);

  x1 = bf2f(qkv[base + CDIM + 2 * j]);
  x2 = bf2f(qkv[base + CDIM + 2 * j + 1]);
  k[qo]     = f2bf(x1 * c - x2 * s);
  k[qo + 1] = f2bf(x1 * s + x2 * c);

  size_t vb = (size_t)(b * HDIM + h) * DDIM;
  vt[(vb + 2 * j)     * TDIM + t] = qkv[base + 2 * CDIM + 2 * j];
  vt[(vb + 2 * j + 1) * TDIM + t] = qkv[base + 2 * CDIM + 2 * j + 1];
}

// ---- Flash attention: grid(T/128, B*H), 8 waves x 16 query rows ----------
#define FSTR 136   // 128 + 8 pad, 272B row stride (16B aligned)
__global__ __launch_bounds__(256, 1)
void flash_attn(const ushort_t* __restrict__ Q, const ushort_t* __restrict__ K,
                const ushort_t* __restrict__ Vt, ushort_t* __restrict__ attv) {
  __shared__ __align__(16) ushort_t Qs[128 * FSTR];
  __shared__ __align__(16) ushort_t Ks[128 * FSTR];
  __shared__ __align__(16) ushort_t Vs[128 * FSTR];
  __shared__ __align__(16) ushort_t Ps[8][16 * FSTR];

  const int tid = threadIdx.x;
  const int wave = tid >> 5, lane = tid & 31;
  const int hl = lane >> 4, ln = lane & 15;
  const int qb = blockIdx.x, bh = blockIdx.y;
  const float scale = 0.08838834764f;  // 1/sqrt(128)

#pragma unroll
  for (int i = 0; i < 8; ++i) {        // Q block 128x128
    int cidx = tid + i * 256;
    int row = cidx >> 4, cc = (cidx & 15) * 8;
    cp16(&Qs[row * FSTR + cc],
         &Q[((size_t)bh * TDIM + qb * 128 + row) * DDIM + cc]);
  }

  float mrow[8], lrow[8];
  v8f Oacc[8];
#pragma unroll
  for (int r = 0; r < 8; ++r) { mrow[r] = -1e30f; lrow[r] = 0.0f; }
#pragma unroll
  for (int dt = 0; dt < 8; ++dt) Oacc[dt] = (v8f)0.0f;

  for (int kb = 0; kb <= qb; ++kb) {
    __syncthreads();                   // everyone done reading prev K/V
#pragma unroll
    for (int i = 0; i < 8; ++i) {
      int cidx = tid + i * 256;
      int row = cidx >> 4, cc = (cidx & 15) * 8;
      cp16(&Ks[row * FSTR + cc],
           &K[((size_t)bh * TDIM + kb * 128 + row) * DDIM + cc]);
      cp16(&Vs[row * FSTR + cc],
           &Vt[((size_t)bh * DDIM + row) * TDIM + kb * 128 + cc]);
    }
    wait_async();
    __syncthreads();

    // S = Q K^T  (16 rows x 128 keys per wave)
    v16bf aq[4];
#pragma unroll
    for (int kk = 0; kk < 4; ++kk)
      aq[kk] = ldfragA(&Qs[(wave * 16 + ln) * FSTR + kk * 32], hl * 8);

    v8f s[8];
#pragma unroll
    for (int nt = 0; nt < 8; ++nt) {
      const ushort_t* krow = &Ks[(nt * 16 + ln) * FSTR];
      v16bf b0 = ldfragB(krow +  0, hl * 16);
      v16bf b1 = ldfragB(krow + 32, hl * 16);
      v16bf b2 = ldfragB(krow + 64, hl * 16);
      v16bf b3 = ldfragB(krow + 96, hl * 16);
      v8f sv = (v8f)0.0f;
      sv = wmma_bf16(aq[0], b0, sv);
      sv = wmma_bf16(aq[1], b1, sv);
      sv = wmma_bf16(aq[2], b2, sv);
      sv = wmma_bf16(aq[3], b3, sv);
#pragma unroll
      for (int r = 0; r < 8; ++r) sv[r] *= scale;
      s[nt] = sv;
    }

    if (kb == qb) {                    // causal mask on diagonal block
#pragma unroll
      for (int nt = 0; nt < 8; ++nt)
#pragma unroll
        for (int r = 0; r < 8; ++r) {
          int key = nt * 16 + ln;
          int qr  = wave * 16 + r + hl * 8;
          if (key > qr) s[nt][r] = -1e30f;
        }
    }

    // online softmax
    float alpha[8];
#pragma unroll
    for (int r = 0; r < 8; ++r) {
      float v = s[0][r];
#pragma unroll
      for (int nt = 1; nt < 8; ++nt) v = fmaxf(v, s[nt][r]);
      v = redmax16(v);
      float nm = fmaxf(mrow[r], v);
      alpha[r] = __expf(mrow[r] - nm);
      mrow[r] = nm;
    }
    float rs[8];
#pragma unroll
    for (int r = 0; r < 8; ++r) rs[r] = 0.0f;
#pragma unroll
    for (int nt = 0; nt < 8; ++nt)
#pragma unroll
      for (int r = 0; r < 8; ++r) {
        float p = __expf(s[nt][r] - mrow[r]);
        rs[r] += p;
        Ps[wave][(r + hl * 8) * FSTR + nt * 16 + ln] = f2bf(p);
      }
#pragma unroll
    for (int r = 0; r < 8; ++r)
      lrow[r] = lrow[r] * alpha[r] + redsum16(rs[r]);
#pragma unroll
    for (int dt = 0; dt < 8; ++dt)
#pragma unroll
      for (int r = 0; r < 8; ++r) Oacc[dt][r] *= alpha[r];

    // O += P V   (per-wave Ps, same-wave LDS ops are in-order)
    v16bf ap[4];
#pragma unroll
    for (int kt = 0; kt < 4; ++kt)
      ap[kt] = ldfragA(&Ps[wave][ln * FSTR + kt * 32], hl * 8);
#pragma unroll
    for (int dt = 0; dt < 8; ++dt) {
      const ushort_t* vrow = &Vs[(dt * 16 + ln) * FSTR];
      v16bf b0 = ldfragB(vrow +  0, hl * 16);
      v16bf b1 = ldfragB(vrow + 32, hl * 16);
      v16bf b2 = ldfragB(vrow + 64, hl * 16);
      v16bf b3 = ldfragB(vrow + 96, hl * 16);
      Oacc[dt] = wmma_bf16(ap[0], b0, Oacc[dt]);
      Oacc[dt] = wmma_bf16(ap[1], b1, Oacc[dt]);
      Oacc[dt] = wmma_bf16(ap[2], b2, Oacc[dt]);
      Oacc[dt] = wmma_bf16(ap[3], b3, Oacc[dt]);
    }
  }

  const int b = bh >> 4, h = bh & 15;
#pragma unroll
  for (int r = 0; r < 8; ++r) {
    float inv = 1.0f / lrow[r];
    int qg = qb * 128 + wave * 16 + r + hl * 8;
#pragma unroll
    for (int dt = 0; dt < 8; ++dt) {
      int dcol = dt * 16 + ln;
      attv[((size_t)(b * TDIM + qg)) * CDIM + h * DDIM + dcol] =
          f2bf(Oacc[dt][r] * inv);
    }
  }
}

// ---- host-side orchestration --------------------------------------------
extern "C" void kernel_launch(void* const* d_in, const int* in_sizes, int n_in,
                              void* d_out, int out_size, void* d_ws, size_t ws_size,
                              hipStream_t stream) {
  const float* x     = (const float*)d_in[0];
  const float* w_qkv = (const float*)d_in[1];
  const float* b_qkv = (const float*)d_in[2];
  const float* w_out = (const float*)d_in[3];
  const float* b_out = (const float*)d_in[4];
  float* out = (float*)d_out;

  char* ws = (char*)d_ws;
  ushort_t* xb    = (ushort_t*)(ws);
  ushort_t* wqkvb = (ushort_t*)(ws + 33554432ull);
  ushort_t* woutb = (ushort_t*)(ws + 58720256ull);
  ushort_t* qkvb  = (ushort_t*)(ws + 67108864ull);
  ushort_t* qbf   = (ushort_t*)(ws + 167772160ull);
  ushort_t* kbf   = (ushort_t*)(ws + 201326592ull);
  ushort_t* vtb   = (ushort_t*)(ws + 234881024ull);
  ushort_t* attv  = xb;  // x dead after QKV GEMM — reuse

  cvt_f32_bf16<<<8192, 256, 0, stream>>>(x, xb, BDIM * TDIM * CDIM / 8);
  cvt_f32_bf16<<<6144, 256, 0, stream>>>(w_qkv, wqkvb, NQKV * CDIM / 8);
  cvt_f32_bf16<<<2048, 256, 0, stream>>>(w_out, woutb, CDIM * CDIM / 8);

  gemm_bf16_nt<true><<<dim3(48, 64), 256, 0, stream>>>(
      xb, wqkvb, b_qkv, qkvb, BDIM * TDIM, NQKV, CDIM);

  rope_rearrange<<<32768, 256, 0, stream>>>(qkvb, qbf, kbf, vtb);

  flash_attn<<<dim3(TDIM / 128, BDIM * HDIM), 256, 0, stream>>>(qbf, kbf, vtb, attv);

  gemm_bf16_nt<false><<<dim3(16, 64), 256, 0, stream>>>(
      attv, woutb, b_out, out, BDIM * TDIM, CDIM, CDIM);
}
